// MultiHeadEdgeAwareMessagePassing_17489106829378
// MI455X (gfx1250) — compile-verified, matrix operand order checked by reading, not run on hardware
//
#include <hip/hip_runtime.h>
#include <hip/hip_bf16.h>

#define NN 3072
#define DM 256
#define NH 4
#define DH_ 64
#define UROW 136   // 2*DH + D_EDGE

typedef __attribute__((ext_vector_type(16))) _Float16 v16h;
typedef __attribute__((ext_vector_type(8)))  _Float16 v8h;
typedef __attribute__((ext_vector_type(8)))  float    v8f;
typedef __attribute__((ext_vector_type(4)))  float    v4f;

// ---------------- WMMA helpers (CDNA5 wave32, 16x16x32 f16 -> f32 acc) ----------------

static __device__ __forceinline__ v8f wmma_f16(v16h a, v16h b, v8f c) {
    // (neg_a, A, neg_b, B, c_mod, C, reuse_a, reuse_b)
    return __builtin_amdgcn_wmma_f32_16x16x32_f16(false, a, false, b, (short)0, c, false, false);
}

// A/B 16-bit fragment layout (ISA 7.12.2): lane l holds 16 K-values for row/col (l&15),
// in two contiguous runs: K = 8g+0..7 and K = 16+8g+0..7, where g = l>>4.
// p1 points at run1 (8 fp32), p2 at run2.
static __device__ __forceinline__ v16h frag_from_f32(const float* p1, const float* p2) {
    v4f x0 = *(const v4f*)(p1);
    v4f x1 = *(const v4f*)(p1 + 4);
    v4f x2 = *(const v4f*)(p2);
    v4f x3 = *(const v4f*)(p2 + 4);
    v16h r;
#pragma unroll
    for (int e = 0; e < 4; ++e) {
        r[e]      = (_Float16)x0[e];
        r[4 + e]  = (_Float16)x1[e];
        r[8 + e]  = (_Float16)x2[e];
        r[12 + e] = (_Float16)x3[e];
    }
    return r;
}

static __device__ __forceinline__ v16h frag_from_f16(const _Float16* p1, const _Float16* p2) {
    v8h x0 = *(const v8h*)(p1);
    v8h x1 = *(const v8h*)(p2);
    v16h r;
#pragma unroll
    for (int e = 0; e < 8; ++e) { r[e] = x0[e]; r[8 + e] = x1[e]; }
    return r;
}

// ---------------- 1) fold u_k into Wk; compute c1, bias_k ----------------

__global__ void prep_kernel(const float* __restrict__ Wk, const float* __restrict__ bk,
                            const float* __restrict__ We_w, const float* __restrict__ u,
                            float* __restrict__ wk_eff, float* __restrict__ c1,
                            float* __restrict__ biask) {
    int t = threadIdx.x;
    for (int e = t; e < NH * DM; e += 256) {
        int hh = e >> 8, c = e & 255;
        float s = 0.f;
#pragma unroll 8
        for (int d = 0; d < DH_; ++d)
            s += u[hh * UROW + DH_ + d] * Wk[(size_t)(hh * DH_ + d) * DM + c];
        wk_eff[e] = s;
    }
    if (t < NH) {
        float s1 = 0.f;
#pragma unroll
        for (int d = 0; d < 8; ++d) s1 += We_w[t * 8 + d] * u[t * UROW + 2 * DH_ + d];
        c1[t] = s1;
        float sb = 0.f;
#pragma unroll 8
        for (int d = 0; d < DH_; ++d) sb += bk[t * DH_ + d] * u[t * UROW + DH_ + d];
        biask[t] = sb;
    }
}

// ---------------- 2) s_kT[h][j] = h[j] . wk_eff[h] + bias_k[h] ----------------

__global__ void sk_kernel(const float* __restrict__ hmat, const float* __restrict__ wk_eff,
                          const float* __restrict__ biask, float* __restrict__ skT) {
    int j = blockIdx.x * 256 + threadIdx.x;
    if (j >= NN) return;
    const float* hr = hmat + (size_t)j * DM;
    float a0 = 0.f, a1 = 0.f, a2 = 0.f, a3 = 0.f;
    for (int c = 0; c < DM; ++c) {
        float hv = hr[c];
        a0 = fmaf(hv, wk_eff[c],           a0);
        a1 = fmaf(hv, wk_eff[DM + c],      a1);
        a2 = fmaf(hv, wk_eff[2 * DM + c],  a2);
        a3 = fmaf(hv, wk_eff[3 * DM + c],  a3);
    }
    skT[0 * NN + j] = a0 + biask[0];
    skT[1 * NN + j] = a1 + biask[1];
    skT[2 * NN + j] = a2 + biask[2];
    skT[3 * NN + j] = a3 + biask[3];
}

// ---------------- 3) v = h @ Wv.T + bv, stored transposed as f16: vT[DM][NN] ----------------

__global__ __launch_bounds__(128) void vgemm_kernel(const float* __restrict__ hmat,
                                                    const float* __restrict__ Wv,
                                                    const float* __restrict__ bv,
                                                    _Float16* __restrict__ vT) {
    int tid = threadIdx.x;
    int wv  = tid >> 5;
    int ln  = tid & 31;
    int g   = ln >> 4;
    int lr  = ln & 15;
    int tile = blockIdx.x * 4 + wv;           // 3072 tiles = 192 i-tiles x 16 m-tiles
    int i0 = (tile >> 4) * 16;
    int m0 = (tile & 15) * 16;

    float bvv = bv[m0 + lr];
    v8f c;
#pragma unroll
    for (int r = 0; r < 8; ++r) c[r] = bvv;

    const float* arow = hmat + (size_t)(i0 + lr) * DM + g * 8;
    const float* brow = Wv   + (size_t)(m0 + lr) * DM + g * 8;
#pragma unroll
    for (int kb = 0; kb < DM; kb += 32) {
        v16h a = frag_from_f32(arow + kb, arow + kb + 16);
        v16h b = frag_from_f32(brow + kb, brow + kb + 16);
        c = wmma_f16(a, b, c);
    }
    // C element [M=r+8g, N=lr] = v[i0+M, m0+lr]  ->  vT[m0+lr][i0+M]
#pragma unroll
    for (int r = 0; r < 8; ++r)
        vT[(size_t)(m0 + lr) * NN + i0 + r + 8 * g] = (_Float16)c[r];
}

// ---------------- 4) per-(row,head) softmax stats; w streamed HBM->L2 once ----------------

__global__ void stats_kernel(const float* __restrict__ w, const float* __restrict__ skT,
                             const float* __restrict__ c1, float* __restrict__ mOut,
                             float* __restrict__ invlOut, int* __restrict__ flagOut) {
    __shared__ float red[256];
    __shared__ int anyN;
    int i = blockIdx.x;
    int t = threadIdx.x;
    const float* wrow = w + (size_t)i * NN;

    float wl[12];
    bool any = false;
#pragma unroll
    for (int k = 0; k < 12; ++k) {
        wl[k] = wrow[t + 256 * k];
        any = any || (wl[k] > 0.f);
    }
    if (t == 0) anyN = 0;
    __syncthreads();
    if (any) atomicOr(&anyN, 1);
    __syncthreads();
    int flag = (anyN == 0) ? 1 : 0;

    for (int h = 0; h < NH; ++h) {
        float c1h = c1[h];
        const float* skr = skT + (size_t)h * NN;
        float mx = -INFINITY;
#pragma unroll
        for (int k = 0; k < 12; ++k)
            if (wl[k] > 0.f) mx = fmaxf(mx, fmaf(c1h, wl[k], skr[t + 256 * k]));
        red[t] = mx;
        __syncthreads();
        for (int s = 128; s > 0; s >>= 1) {
            if (t < s) red[t] = fmaxf(red[t], red[t + s]);
            __syncthreads();
        }
        float m = red[0];
        __syncthreads();

        float se = 0.f;
#pragma unroll
        for (int k = 0; k < 12; ++k)
            if (wl[k] > 0.f) se += __expf(fmaf(c1h, wl[k], skr[t + 256 * k]) - m);
        red[t] = se;
        __syncthreads();
        for (int s = 128; s > 0; s >>= 1) {
            if (t < s) red[t] += red[t + s];
            __syncthreads();
        }
        float l = red[0];
        __syncthreads();

        if (t == 0) {
            if (flag) { mOut[i * NH + h] = 0.f;  invlOut[i * NH + h] = 1.0f / (float)NN; }
            else      { mOut[i * NH + h] = m;    invlOut[i * NH + h] = 1.0f / l; }
        }
    }
    if (t == 0) flagOut[i] = flag;
}

// ---------------- 5) msg = (softmax(logits)*w) @ v  via f16 WMMA, fp32 acc ----------------
// Block: 16 rows, 4 waves = 4 heads. Second pass over w hits L2 (192MB holds all of w).

__global__ __launch_bounds__(128) void attn_kernel(const float* __restrict__ w,
                                                   const float* __restrict__ skT,
                                                   const float* __restrict__ c1,
                                                   const float* __restrict__ mIn,
                                                   const float* __restrict__ invlIn,
                                                   const int* __restrict__ flags,
                                                   const _Float16* __restrict__ vT,
                                                   _Float16* __restrict__ msg) {
    int tid = threadIdx.x;
    int h   = tid >> 5;        // wave == head
    int ln  = tid & 31;
    int g   = ln >> 4;
    int lr  = ln & 15;
    int i0  = blockIdx.x * 16;
    int iL  = i0 + lr;

    float mrow = mIn[iL * NH + h];
    float invl = invlIn[iL * NH + h];
    int   flag = flags[iL];
    float c1h  = c1[h];

    v8f acc[4];
#pragma unroll
    for (int dt = 0; dt < 4; ++dt)
#pragma unroll
        for (int r = 0; r < 8; ++r) acc[dt][r] = 0.f;

    const float*    wrow  = w   + (size_t)iL * NN + g * 8;
    const float*    skr   = skT + (size_t)h * NN + g * 8;
    const _Float16* vbase = vT  + (size_t)(h * DH_ + lr) * NN + g * 8;

    for (int j0 = 0; j0 < NN; j0 += 32) {
        v4f w0 = *(const v4f*)(wrow + j0);
        v4f w1 = *(const v4f*)(wrow + j0 + 4);
        v4f w2 = *(const v4f*)(wrow + j0 + 16);
        v4f w3 = *(const v4f*)(wrow + j0 + 20);
        v4f s0 = *(const v4f*)(skr + j0);
        v4f s1 = *(const v4f*)(skr + j0 + 4);
        v4f s2 = *(const v4f*)(skr + j0 + 16);
        v4f s3 = *(const v4f*)(skr + j0 + 20);

        v16h a;
#pragma unroll
        for (int e = 0; e < 4; ++e) {
            float wv0 = w0[e], wv1 = w1[e], wv2 = w2[e], wv3 = w3[e];
            float p0 = flag ? invl * wv0
                            : ((wv0 > 0.f) ? invl * wv0 * __expf(fmaf(c1h, wv0, s0[e]) - mrow) : 0.f);
            float p1 = flag ? invl * wv1
                            : ((wv1 > 0.f) ? invl * wv1 * __expf(fmaf(c1h, wv1, s1[e]) - mrow) : 0.f);
            float p2 = flag ? invl * wv2
                            : ((wv2 > 0.f) ? invl * wv2 * __expf(fmaf(c1h, wv2, s2[e]) - mrow) : 0.f);
            float p3 = flag ? invl * wv3
                            : ((wv3 > 0.f) ? invl * wv3 * __expf(fmaf(c1h, wv3, s3[e]) - mrow) : 0.f);
            a[e]      = (_Float16)p0;
            a[4 + e]  = (_Float16)p1;
            a[8 + e]  = (_Float16)p2;
            a[12 + e] = (_Float16)p3;
        }

#pragma unroll
        for (int dt = 0; dt < 4; ++dt) {
            const _Float16* vp = vbase + (size_t)(dt * 16) * NN + j0;
            v16h b = frag_from_f16(vp, vp + 16);
            acc[dt] = wmma_f16(a, b, acc[dt]);
        }
    }

    // C element [M=r+8g, N=lr] of head h, dtile dt -> msg[i0+M][h*64 + dt*16 + lr]
#pragma unroll
    for (int dt = 0; dt < 4; ++dt)
#pragma unroll
        for (int r = 0; r < 8; ++r)
            msg[(size_t)(i0 + r + 8 * g) * DM + h * DH_ + dt * 16 + lr] = (_Float16)acc[dt][r];
}

// ---------------- 6) out = msg @ Wo.T + bo; x = h + out; LayerNorm ----------------
// Block covers 16 rows x full 256 cols (4 waves x 4 m-tiles) so LN reduces in-block.

__global__ __launch_bounds__(128) void outln_kernel(const _Float16* __restrict__ msg,
                                                    const float* __restrict__ Wo,
                                                    const float* __restrict__ bo,
                                                    const float* __restrict__ hin,
                                                    const float* __restrict__ gamma,
                                                    const float* __restrict__ beta,
                                                    float* __restrict__ out) {
    __shared__ float rs[16];
    __shared__ float rss[16];
    int tid = threadIdx.x;
    int wv  = tid >> 5;
    int ln  = tid & 31;
    int g   = ln >> 4;
    int lr  = ln & 15;
    int i0  = blockIdx.x * 16;

    v8f acc[4];
#pragma unroll
    for (int q = 0; q < 4; ++q) {
        float bvv = bo[(wv * 4 + q) * 16 + lr];
#pragma unroll
        for (int r = 0; r < 8; ++r) acc[q][r] = bvv;
    }

    const _Float16* arow = msg + (size_t)(i0 + lr) * DM + g * 8;
#pragma unroll
    for (int kb = 0; kb < DM; kb += 32) {
        v16h a = frag_from_f16(arow + kb, arow + kb + 16);
#pragma unroll
        for (int q = 0; q < 4; ++q) {
            const float* brow = Wo + (size_t)((wv * 4 + q) * 16 + lr) * DM + kb + g * 8;
            v16h b = frag_from_f32(brow, brow + 16);
            acc[q] = wmma_f16(a, b, acc[q]);
        }
    }

    if (tid < 16) { rs[tid] = 0.f; rss[tid] = 0.f; }
    __syncthreads();

    float xv[4][8];
    float ps1[8], ps2[8];
#pragma unroll
    for (int r = 0; r < 8; ++r) { ps1[r] = 0.f; ps2[r] = 0.f; }
#pragma unroll
    for (int q = 0; q < 4; ++q) {
        int col = (wv * 4 + q) * 16 + lr;
#pragma unroll
        for (int r = 0; r < 8; ++r) {
            int row = i0 + r + 8 * g;
            float x = acc[q][r] + hin[(size_t)row * DM + col];
            xv[q][r] = x;
            ps1[r] += x;
            ps2[r] += x * x;
        }
    }
#pragma unroll
    for (int r = 0; r < 8; ++r) {
        atomicAdd(&rs[r + 8 * g], ps1[r]);     // ds_add_f32
        atomicAdd(&rss[r + 8 * g], ps2[r]);
    }
    __syncthreads();

    const float invd = 1.0f / (float)DM;
#pragma unroll
    for (int q = 0; q < 4; ++q) {
        int col = (wv * 4 + q) * 16 + lr;
        float gma = gamma[col], bta = beta[col];
#pragma unroll
        for (int r = 0; r < 8; ++r) {
            int rl = r + 8 * g;
            float mu  = rs[rl] * invd;
            float var = rss[rl] * invd - mu * mu;
            float inv = rsqrtf(var + 1e-5f);
            out[(size_t)(i0 + rl) * DM + col] = gma * (xv[q][r] - mu) * inv + bta;
        }
    }
}

// ---------------- launch ----------------

extern "C" void kernel_launch(void* const* d_in, const int* in_sizes, int n_in,
                              void* d_out, int out_size, void* d_ws, size_t ws_size,
                              hipStream_t stream) {
    (void)in_sizes; (void)n_in; (void)out_size; (void)ws_size;

    const float* h_in = (const float*)d_in[0];
    const float* w    = (const float*)d_in[1];
    // Wq (2), bq (3) cancel in the softmax -> unused.
    const float* Wk   = (const float*)d_in[4];
    const float* bk   = (const float*)d_in[5];
    const float* Wv   = (const float*)d_in[6];
    const float* bv   = (const float*)d_in[7];
    const float* We_w = (const float*)d_in[8];
    // We_b (9) gives c0: row-constant in softmax -> unused.
    const float* u    = (const float*)d_in[10];
    const float* Wo   = (const float*)d_in[11];
    const float* bo   = (const float*)d_in[12];
    const float* gmma = (const float*)d_in[13];
    const float* beta = (const float*)d_in[14];
    float* out = (float*)d_out;

    char* ws = (char*)d_ws;
    float*    wk_eff = (float*)(ws + 0);                 //  4096 B
    float*    c1     = (float*)(ws + 4096);              //    16 B
    float*    biask  = (float*)(ws + 4112);              //    16 B
    float*    skT    = (float*)(ws + 4352);              // 49152 B  [H][N]
    float*    mbuf   = (float*)(ws + 53504);             // 49152 B  [N][H]
    float*    invl   = (float*)(ws + 102656);            // 49152 B  [N][H]
    int*      flags  = (int*)  (ws + 151808);            // 12288 B  [N]
    _Float16* vT     = (_Float16*)(ws + 164096);         // 1572864 B [DM][N]
    _Float16* msg    = (_Float16*)(ws + 1736960);        // 1572864 B [N][DM]
    // total: 3,309,824 B

    prep_kernel<<<1, 256, 0, stream>>>(Wk, bk, We_w, u, wk_eff, c1, biask);
    sk_kernel<<<NN / 256, 256, 0, stream>>>(h_in, wk_eff, biask, skT);
    vgemm_kernel<<<(NN / 16) * (DM / 16) / 4, 128, 0, stream>>>(h_in, Wv, bv, vT);
    stats_kernel<<<NN, 256, 0, stream>>>(w, skT, c1, mbuf, invl, flags);
    attn_kernel<<<NN / 16, 128, 0, stream>>>(w, skT, c1, mbuf, invl, flags, vT, msg);
    outln_kernel<<<NN / 16, 128, 0, stream>>>(msg, Wo, bo, h_in, gmma, beta, out);
}
